// MultigpuGCNConv_87960930222589
// MI455X (gfx1250) — matrix operand order repaired and checked by reading, not compiled, not run on hardware
//
#include <hip/hip_runtime.h>

// ---------------------------------------------------------------------------
// GCN layer: out = relu( A_csr @ (in_feat @ W) + bias )
//   in_feat: [50000 x 256] f32, W: [256 x 128] f32, DEG = 16 edges/row.
// Pass 0: swizzle W -> bf16 in B-fragment lane order (once, 64KB, L2-resident)
// Pass 1: h = bf16(in_feat) @ bf16(W) via v_wmma_f32_16x16x32_bf16, f32 accum
// Pass 2: per-row SpMM gather + bias + relu (bandwidth bound, h L2-resident)
// ---------------------------------------------------------------------------

#define N_NODES 50000
#define IN_F    256
#define OUT_F   128
#define M_TILES (N_NODES / 16)     // 3125 (exact)
#define K_CHUNKS (IN_F / 32)       // 8
#define N_TILES  (OUT_F / 16)      // 8

typedef __attribute__((ext_vector_type(16))) __bf16 v16bf;
typedef __attribute__((ext_vector_type(8)))  float  v8f;

static __device__ __forceinline__ unsigned f2bf_hi(float f) {
  // round-to-nearest-even f32 -> bf16 bits in [15:0]
  unsigned u = __float_as_uint(f);
  u += 0x7FFFu + ((u >> 16) & 1u);
  return u >> 16;
}

// Fragment K mapping used here for 16-bit A and B: lane-half h, element j ->
//   K_local = j + 16*h         (j in 0..15)
// Inverse (K_local kl -> half, j):  half = kl>>4 ; j = kl&15
// Consecutive even/odd kl map to the same lane and adjacent j -> pack as b32.

// ---------------------------------------------------------------------------
// Pass 0: Wswz[((ntile*8 + kchunk)*32 + lane)*16 + j] = bf16(W[k][n])
//   B fragment (32x16): VGPR v holds K=2v,2v+1; lane-half splits K 0-15/16-31
//   k = kchunk*32 + j + 16*(lane>>4),  n = ntile*16 + (lane&15)
// 16384 packed uints total; grid 64 x 256.
// ---------------------------------------------------------------------------
__global__ void __launch_bounds__(256)
gcn_swizzle_W(const float* __restrict__ W, unsigned* __restrict__ Wswz)
{
  const unsigned u = blockIdx.x * 256 + threadIdx.x;   // 0..16383
  const int j    = (u & 7) << 1;                       // even j
  const int lane = (u >> 3) & 31;
  const int kk   = (u >> 8) & 7;
  const int nt   = u >> 11;
  const int half = lane >> 4;
  const int n    = nt * 16 + (lane & 15);
  const int k    = kk * 32 + j + 16 * half;
  const unsigned lo = f2bf_hi(W[k * OUT_F + n]);
  const unsigned hi = f2bf_hi(W[(k + 1) * OUT_F + n]);
  Wswz[u] = lo | (hi << 16);
}

// ---------------------------------------------------------------------------
// Pass 1: H[50000x128] = in_feat[50000x256] @ W[256x128]
// grid.x = 3125 (one 16-row strip per block), block = 256 = 8 waves.
// A strip staged in LDS pre-swizzled to fragment order; all 8 waves reuse it.
// Wave w owns columns [16w,16w+16); inner loop = 2 ds_b128 + 2 global_b128 + wmma.
// ---------------------------------------------------------------------------
__global__ void __launch_bounds__(256)
gcn_gemm_bf16_wmma(const float* __restrict__ A,        // in_feat
                   const unsigned* __restrict__ Wswz,  // pre-swizzled bf16 W
                   float* __restrict__ H)
{
  // [kchunk][lane][16 bf16] = 8*32*8 packed uints = 8 KB
  __shared__ __align__(32) unsigned sAswz[K_CHUNKS * 32 * 8];

  const int tid   = threadIdx.x;
  const int mBase = blockIdx.x * 16;

  // Stage + convert + swizzle the 16x256 A strip (2048 packed-pair elements).
  #pragma unroll
  for (int i = tid; i < 16 * (IN_F / 2); i += 256) {
    const int r  = i >> 7;             // strip row 0..15
    const int c  = (i & 127) << 1;     // even column 0..254
    const int kk = c >> 5;
    const int kl = c & 31;
    const int hf = kl >> 4;
    const int j  = kl & 15;
    const int ln = (hf << 4) | r;
    const unsigned lo = f2bf_hi(A[(mBase + r) * IN_F + c]);
    const unsigned hi = f2bf_hi(A[(mBase + r) * IN_F + c + 1]);
    sAswz[(kk * 32 + ln) * 8 + (j >> 1)] = lo | (hi << 16);
  }
  __syncthreads();

  const int wave = tid >> 5;   // 0..7 -> N tile
  const int lane = tid & 31;

  const v16bf* aFrag = (const v16bf*)sAswz;                 // [kk*32 + lane]
  const v16bf* bFrag = (const v16bf*)Wswz + wave * (K_CHUNKS * 32);

  v8f c = {};
  #pragma unroll
  for (int kk = 0; kk < K_CHUNKS; ++kk) {
    const v16bf a = aFrag[kk * 32 + lane];
    const v16bf b = bFrag[kk * 32 + lane];
    c = __builtin_amdgcn_wmma_f32_16x16x32_bf16(
        /*neg_a=*/false, a, /*neg_b=*/false, b,
        /*c_mod=*/(short)0, c, /*reuse_a=*/false, /*reuse_b=*/false);
  }

  // C/D layout: VGPR r -> M = r + 8*(lane>>4), N = lane&15
  const int half = lane >> 4;
  const int n    = wave * 16 + (lane & 15);
  #pragma unroll
  for (int r = 0; r < 8; ++r) {
    const int m = mBase + r + half * 8;
    H[m * OUT_F + n] = c[r];
  }
}

// ---------------------------------------------------------------------------
// Pass 2: out[i,:] = relu( sum_e val[e] * h[col[e],:] + bias )
// One block (128 threads) per row: edge metadata is block-uniform (scalar
// loads); h-row gathers are 512B coalesced and hit L2 (h = 25.6 MB < 192 MB).
// ---------------------------------------------------------------------------
__global__ void __launch_bounds__(128)
gcn_spmm_bias_relu(const int*   __restrict__ row_ptr,
                   const int*   __restrict__ col_idx,
                   const float* __restrict__ eval,
                   const float* __restrict__ h,
                   const float* __restrict__ bias,
                   float*       __restrict__ out)
{
  const int row = blockIdx.x;
  const int t   = threadIdx.x;           // feature 0..127

  const int s = row_ptr[row];
  const int e = row_ptr[row + 1];

  float acc = bias[t];
  #pragma unroll 4
  for (int i = s; i < e; ++i) {
    const int   c = col_idx[i];
    const float w = eval[i];
    acc = fmaf(w, h[(long)c * OUT_F + t], acc);
  }
  out[(long)row * OUT_F + t] = fmaxf(acc, 0.0f);
}

// ---------------------------------------------------------------------------
// Launch. Input order (setup_inputs dict):
//  0:nnz 1:row_ptr_s 2:col_idx_s 3:edge_val_csr_s 4:p_csr 5:q_csr
//  6:col_ptr_s 7:row_idx_s 8:edge_val_csc_s 9:p_csc 10:q_csc
//  11:in_feat 12:W 13:bias
// d_ws layout: [0, 25.6MB) = h (f32), [25.6MB, +64KB) = Wswz (bf16 packed)
// ---------------------------------------------------------------------------
extern "C" void kernel_launch(void* const* d_in, const int* in_sizes, int n_in,
                              void* d_out, int out_size, void* d_ws, size_t ws_size,
                              hipStream_t stream) {
  (void)in_sizes; (void)n_in; (void)out_size; (void)ws_size;

  const int*   row_ptr = (const int*)  d_in[1];
  const int*   col_idx = (const int*)  d_in[2];
  const float* eval    = (const float*)d_in[3];
  const float* in_feat = (const float*)d_in[11];
  const float* W       = (const float*)d_in[12];
  const float* bias    = (const float*)d_in[13];

  const size_t hBytes = (size_t)N_NODES * OUT_F * sizeof(float);  // 25,600,000 (32-aligned)
  float*    h    = (float*)d_ws;
  unsigned* Wswz = (unsigned*)((char*)d_ws + hBytes);
  float*    out  = (float*)d_out;

  gcn_swizzle_W     <<<dim3(64),      dim3(256), 0, stream>>>(W, Wswz);
  gcn_gemm_bf16_wmma<<<dim3(M_TILES), dim3(256), 0, stream>>>(in_feat, Wswz, h);
  gcn_spmm_bias_relu<<<dim3(N_NODES), dim3(128), 0, stream>>>(row_ptr, col_idx, eval,
                                                              h, bias, out);
}
